// WeakLayer_16999480557594
// MI455X (gfx1250) — matrix operand order repaired and checked by reading, not compiled, not run on hardware
//
#include <hip/hip_runtime.h>
#include <hip/hip_bf16.h>
#include <math.h>

typedef __attribute__((ext_vector_type(16))) _Float16 v16h;
typedef __attribute__((ext_vector_type(8)))  float    v8f;

#define NPTS 32768
#define CH   64
#define KN   16
#define ROWV 68   // f32 LDS row stride for sV (floats); 272B = 17*16B, keeps 16B alignment

// ---------------------------------------------------------------------------
// K0: zero the w_d accumulator (C*K = 1024 floats)
// ---------------------------------------------------------------------------
__global__ void wl_zero(float* __restrict__ wd) {
  for (int i = threadIdx.x; i < CH * KN; i += blockDim.x) wd[i] = 0.0f;
}

// ---------------------------------------------------------------------------
// K1: WMMA projections  x_q = xt@Wq+bq, y_k = yt@Wk+bk, y_v = yt@Wv+bv
// One wave per block; 16-row x 64-col tile of all three per block.
// ---------------------------------------------------------------------------
__global__ __launch_bounds__(32) void wl_proj(
    const float* __restrict__ x, const float* __restrict__ y,
    const float* __restrict__ Wq, const float* __restrict__ bq,
    const float* __restrict__ Wk, const float* __restrict__ bk,
    const float* __restrict__ Wv, const float* __restrict__ bv,
    float* __restrict__ xq, float* __restrict__ yk, float* __restrict__ yv) {
  const int n0   = blockIdx.x * 16;
  const int lane = threadIdx.x & 31;
  const int m16  = lane & 15;
  const int hi   = lane >> 4;

  v16h ax[2], ay[2];
  #pragma unroll
  for (int kc = 0; kc < 2; ++kc) {
    #pragma unroll
    for (int e = 0; e < 16; ++e) {
      int K = kc * 32 + (e & 7) + ((e >> 3) << 4) + (hi << 3);   // A layout
      ax[kc][e] = (_Float16)x[K * NPTS + n0 + m16];
      ay[kc][e] = (_Float16)y[K * NPTS + n0 + m16];
    }
  }

  #pragma unroll
  for (int nt = 0; nt < 4; ++nt) {
    const int col = nt * 16 + m16;
    v8f aq = {}, ak = {}, av = {};
    #pragma unroll
    for (int kc = 0; kc < 2; ++kc) {
      v16h bqm, bkm, bvm;
      #pragma unroll
      for (int e = 0; e < 16; ++e) {
        int K = kc * 32 + (e & 7) + ((e >> 3) << 4) + (hi << 3);  // B layout
        bqm[e] = (_Float16)Wq[K * CH + col];
        bkm[e] = (_Float16)Wk[K * CH + col];
        bvm[e] = (_Float16)Wv[K * CH + col];
      }
      aq = __builtin_amdgcn_wmma_f32_16x16x32_f16(false, ax[kc], false, bqm, (short)0, aq, false, false);
      ak = __builtin_amdgcn_wmma_f32_16x16x32_f16(false, ay[kc], false, bkm, (short)0, ak, false, false);
      av = __builtin_amdgcn_wmma_f32_16x16x32_f16(false, ay[kc], false, bvm, (short)0, av, false, false);
    }
    const float bqc = bq[col], bkc = bk[col], bvc = bv[col];
    #pragma unroll
    for (int j = 0; j < 8; ++j) {
      const int row = n0 + j + (hi << 3);                         // D layout
      xq[row * CH + col] = aq[j] + bqc;
      yk[row * CH + col] = ak[j] + bkc;
      yv[row * CH + col] = av[j] + bvc;
    }
  }
}

// ---------------------------------------------------------------------------
// K2: x_uni[r] = relu(bn(x.flat[r*64:+64] . Wu + bu))   (reshape, NOT transpose)
// ---------------------------------------------------------------------------
__global__ void wl_uni(const float* __restrict__ x, const float* __restrict__ Wu,
                       const float* __restrict__ bu, const float* __restrict__ gu,
                       const float* __restrict__ bnu, float* __restrict__ xuni) {
  const int r = blockIdx.x * blockDim.x + threadIdx.x;
  if (r >= NPTS) return;
  const float* px = x + (size_t)r * CH;
  float s = 0.0f;
  #pragma unroll 8
  for (int j = 0; j < CH; ++j) s = fmaf(px[j], Wu[j], s);
  s += bu[0];
  s = s * (gu[0] * rsqrtf(1.0f + 1e-5f)) + bnu[0];
  xuni[r] = fmaxf(s, 0.0f);
}

// ---------------------------------------------------------------------------
// K3: wd[c][k] += sum_n (y_k[idx[n,k]][c] - x_q[n][c]) * x_uni[n]
// ---------------------------------------------------------------------------
__global__ __launch_bounds__(256) void wl_wd(
    const float* __restrict__ xq, const float* __restrict__ yk,
    const float* __restrict__ xuni, const int* __restrict__ idx,
    float* __restrict__ wd) {
  __shared__ float part[4][CH][KN + 1];
  const int t = threadIdx.x;
  const int c = t & 63, g = t >> 6;
  float acc[KN];
  #pragma unroll
  for (int k = 0; k < KN; ++k) acc[k] = 0.0f;
  const int base = blockIdx.x * 128;
  for (int i = 0; i < 32; ++i) {
    const int n = base + g + 4 * i;
    const float xu  = xuni[n];
    const float xqc = xq[(size_t)n * CH + c];
    const int* pidx = idx + (size_t)n * KN;
    #pragma unroll
    for (int k = 0; k < KN; ++k) {
      const int gi = pidx[k];
      acc[k] = fmaf(yk[(size_t)gi * CH + c] - xqc, xu, acc[k]);
    }
  }
  #pragma unroll
  for (int k = 0; k < KN; ++k) part[g][c][k] = acc[k];
  __syncthreads();
  if (g == 0) {
    #pragma unroll
    for (int k = 0; k < KN; ++k) {
      float s = part[0][c][k] + part[1][c][k] + part[2][c][k] + part[3][c][k];
      atomicAdd(&wd[c * KN + k], s);
    }
  }
}

// ---------------------------------------------------------------------------
// K4: softmax w_d over K axis (per channel row), in place
// ---------------------------------------------------------------------------
__global__ void wl_wdsm(float* __restrict__ wd) {
  const int c = threadIdx.x;
  if (c >= CH) return;
  float v[KN], m = -1e30f;
  #pragma unroll
  for (int k = 0; k < KN; ++k) { v[k] = wd[c * KN + k]; m = fmaxf(m, v[k]); }
  float s = 0.0f;
  #pragma unroll
  for (int k = 0; k < KN; ++k) { v[k] = __expf(v[k] - m); s += v[k]; }
  const float inv = 1.0f / s;
  #pragma unroll
  for (int k = 0; k < KN; ++k) wd[c * KN + k] = v[k] * inv;
}

// WMMA fragment-chunk offsets: each chunk = 16 contiguous halves (32 B) = one
// lane's worth of an A or B operand for one (tile, k-chunk, lane-half).
__device__ __forceinline__ int afrag(int mt, int kc, int hi, int m16) {
  return ((((mt * 2 + kc) * 2 + hi) * 16) + m16) * 16;
}
__device__ __forceinline__ int wfrag(int nt, int kc, int hi, int m16) {
  return ((((nt * 2 + kc) * 2 + hi) * 16) + m16) * 16;
}

__device__ __forceinline__ void async_ld_b128(unsigned lds, unsigned goff,
                                              const void* sbase) {
  asm volatile("global_load_async_to_lds_b128 %0, %1, %2 offset:0"
               :: "v"(lds), "v"(goff), "s"(sbase) : "memory");
}
__device__ __forceinline__ void wait_async0() {
#if __has_builtin(__builtin_amdgcn_s_wait_asynccnt)
  __builtin_amdgcn_s_wait_asynccnt(0);
#else
  asm volatile("s_wait_asynccnt 0x0" ::: "memory");
#endif
}

// ---------------------------------------------------------------------------
// K5: fused main kernel. 16 points per block, 256 threads (8 waves).
//  - y_v rows async-DMA'd straight into the sV LDS tile (ASYNCcnt fence)
//  - w_e scores + softmax; geometry MLPs accumulate p_r into sV in place
//  - w0 -> bn/relu -> f16 packed WMMA fragments -> GEMM1 (Ww1) -> bn/relu
//  - GEMM2 (Ww2) -> softmax over k via lane-pair shuffle -> combine -> out
// ---------------------------------------------------------------------------
__global__ __launch_bounds__(256) void wl_fused(
    const float* __restrict__ y, const float* __restrict__ y_xyz,
    const int* __restrict__ idx, const float* __restrict__ yv,
    const float* __restrict__ wd,
    const float* __restrict__ Wp1, const float* __restrict__ bp1,
    const float* __restrict__ gp,  const float* __restrict__ bnp,
    const float* __restrict__ Wp2, const float* __restrict__ bp2,
    const float* __restrict__ Wa1, const float* __restrict__ ba1,
    const float* __restrict__ ga,  const float* __restrict__ bna,
    const float* __restrict__ Wa2, const float* __restrict__ ba2,
    const float* __restrict__ gw1, const float* __restrict__ bw1b,
    const float* __restrict__ Ww1, const float* __restrict__ bw1,
    const float* __restrict__ gw2, const float* __restrict__ bw2b,
    const float* __restrict__ Ww2, const float* __restrict__ bw2,
    const float* __restrict__ a_p, const float* __restrict__ b_p,
    const float* __restrict__ d_p, float* __restrict__ out) {
  __shared__ __align__(32) float    sV[256 * ROWV];        // y_vg + p_r (f32)
  __shared__ __align__(32) _Float16 sA[16 * 2 * 2 * 16 * 16]; // GEMM1 A frags
  __shared__ __align__(32) _Float16 sB[16 * 2 * 2 * 16 * 16]; // GEMM2 A frags
  __shared__ __align__(32) _Float16 sW[4 * 2 * 2 * 16 * 16];  // B (weight) frags
  __shared__ float sWD[CH * KN];
  __shared__ float sS[16][KN];
  __shared__ float sWE[16][KN];
  __shared__ float sVecs[8 * CH];   // Wp2 rows 0-2, Wa2 rows 0-2, bp2, ba2
  __shared__ float sBN[6 * CH];     // gw1*rs, bw1b, bw1, gw2*rs, bw2b, bw2

  const int t  = threadIdx.x;
  const int n0 = blockIdx.x * 16;
  const int p  = t >> 4, k = t & 15;
  const int n  = n0 + p;
  const int g  = idx[(size_t)n * KN + k];
  const float RS = rsqrtf(1.0f + 1e-5f);

  // ---- async DMA: y_v[g] row (256 B) straight into this (p,k)'s sV row ----
  {
    const unsigned lds  = (unsigned)(uintptr_t)&sV[(p * 16 + k) * ROWV];
    const unsigned goff = (unsigned)g * (CH * 4);
    #pragma unroll
    for (int i = 0; i < 16; ++i)
      async_ld_b128(lds + i * 16, goff + i * 16, yv);
  }
  __builtin_prefetch(Ww1 + t * 4, 0, 1);
  __builtin_prefetch(Ww2 + t * 4, 0, 1);

  for (int i = t; i < CH * KN; i += 256) sWD[i] = wd[i];
  if (t < CH) {
    #pragma unroll
    for (int j = 0; j < 3; ++j) {
      sVecs[j * CH + t]       = Wp2[j * CH + t];
      sVecs[(3 + j) * CH + t] = Wa2[j * CH + t];
    }
    sVecs[6 * CH + t] = bp2[t];
    sVecs[7 * CH + t] = ba2[t];
    sBN[0 * CH + t] = gw1[t] * RS;
    sBN[1 * CH + t] = bw1b[t];
    sBN[2 * CH + t] = bw1[t];
    sBN[3 * CH + t] = gw2[t] * RS;
    sBN[4 * CH + t] = bw2b[t];
    sBN[5 * CH + t] = bw2[t];
  }
  __syncthreads();

  // ---- w_e scores: s[p][k] = sum_c y_r[n][c] * w_d[c][k] ----
  {
    float s = 0.0f;
    #pragma unroll 8
    for (int c = 0; c < CH; ++c) s = fmaf(y[c * NPTS + n], sWD[c * KN + k], s);
    sS[p][k] = s;
  }
  __syncthreads();
  if (t < 16) {                                    // softmax over k per point
    float m = -1e30f;
    #pragma unroll
    for (int kk = 0; kk < KN; ++kk) m = fmaxf(m, sS[t][kk]);
    float s = 0.0f;
    #pragma unroll
    for (int kk = 0; kk < KN; ++kk) { float e = __expf(sS[t][kk] - m); sWE[t][kk] = e; s += e; }
    const float inv = 1.0f / s;
    #pragma unroll
    for (int kk = 0; kk < KN; ++kk) sWE[t][kk] *= inv;
  }

  // ---- geometry MLPs (overlapped with async DMA in flight) ----
  float a3[3], q3[3];
  {
    const float d0 = y_xyz[g]            - y_xyz[n];
    const float d1 = y_xyz[NPTS + g]     - y_xyz[NPTS + n];
    const float d2 = y_xyz[2 * NPTS + g] - y_xyz[2 * NPTS + n];
    const float ed  = sqrtf(d0 * d0 + d1 * d1 + d2 * d2);
    const float exy = sqrtf(d0 * d0 + d1 * d1);
    const float cos_e = (ed  > 0.0f) ? exy / ed  : 0.0f;   // nan_to_num(0/0)
    const float cos_a = (exy > 0.0f) ? d1  / exy : 0.0f;
    #pragma unroll
    for (int j = 0; j < 3; ++j) {
      float v = ed * Wa1[j] + cos_a * Wa1[3 + j] + cos_e * Wa1[6 + j] + ba1[j];
      v = v * (ga[j] * RS) + bna[j];
      a3[j] = fmaxf(v, 0.0f);
      float u = d0 * Wp1[j] + d1 * Wp1[3 + j] + d2 * Wp1[6 + j] + bp1[j];
      u = u * (gp[j] * RS) + bnp[j];
      q3[j] = fmaxf(u, 0.0f);
    }
  }

  // ---- fence this wave's DMAs, then accumulate p_r into sV in place ----
  wait_async0();
  {
    const float* dprow = d_p + k * CH;
    float* dst = &sV[(p * 16 + k) * ROWV];
    #pragma unroll 4
    for (int c = 0; c < CH; ++c) {
      float pr = fmaf(q3[0], sVecs[c], fmaf(q3[1], sVecs[CH + c],
                 fmaf(q3[2], sVecs[2 * CH + c], sVecs[6 * CH + c])));
      float af = fmaf(a3[0], sVecs[3 * CH + c], fmaf(a3[1], sVecs[4 * CH + c],
                 fmaf(a3[2], sVecs[5 * CH + c], sVecs[7 * CH + c])));
      dst[c] += pr + dprow[c] * af;
    }
  }

  // ---- stage Ww1 as packed B fragments ----
  for (int i = t; i < CH * CH; i += 256) {
    const int kk = i >> 6, cc = i & 63;
    sW[wfrag(cc >> 4, kk >> 5, (kk >> 3) & 1, cc & 15) + ((kk & 7) | (((kk >> 4) & 1) << 3))] = (_Float16)Ww1[i];
  }
  __syncthreads();   // sWE, sV, sW all ready

  // ---- w0 -> bn/relu -> packed A fragments (32 B vector stores) ----
  {
    const float apk = a_p[k];
    const float we  = sWE[p][k];
    const float* bprow = b_p + k * CH;
    #pragma unroll
    for (int kc = 0; kc < 2; ++kc) {
      #pragma unroll
      for (int h2 = 0; h2 < 2; ++h2) {
        v16h pack;
        #pragma unroll
        for (int e = 0; e < 16; ++e) {
          const int c = kc * 32 + 8 * h2 + e + ((e >= 8) ? 8 : 0);
          float w0 = fmaf(apk, sWD[c * KN + k], bprow[c] * we);
          float h  = fmaxf(fmaf(w0, sBN[c], sBN[CH + c]), 0.0f);
          pack[e] = (_Float16)h;
        }
        *(v16h*)&sA[afrag(p, kc, h2, k)] = pack;
      }
    }
  }
  __syncthreads();

  const int lane = t & 31, wv32 = t >> 5;
  const int m16 = lane & 15, hi = lane >> 4;

  // ---- GEMM1: sA @ Ww1 -> bn/relu -> sB fragments ----
  #pragma unroll
  for (int i = 0; i < 8; ++i) {
    const int T = wv32 * 8 + i;
    const int mt = T >> 2, nt = T & 3;
    const int col = nt * 16 + m16;
    v8f acc = {};
    #pragma unroll
    for (int kc = 0; kc < 2; ++kc) {
      const v16h a = *(const v16h*)&sA[afrag(mt, kc, hi, m16)];
      const v16h b = *(const v16h*)&sW[wfrag(nt, kc, hi, m16)];
      acc = __builtin_amdgcn_wmma_f32_16x16x32_f16(false, a, false, b, (short)0, acc, false, false);
    }
    // scatter into GEMM2 A-fragment slots: (kc2,hi2,e2) fixed per output column
    const int kc2 = col >> 5;
    const int hi2 = (col >> 3) & 1;
    const int e2  = (col & 7) | (((col >> 4) & 1) << 3);
    const float b1 = sBN[2 * CH + col];
    const float sc = sBN[3 * CH + col];
    const float sh = sBN[4 * CH + col];
    #pragma unroll
    for (int j = 0; j < 8; ++j) {
      const int m2 = j + (hi << 3);
      float h = fmaxf(fmaf(acc[j] + b1, sc, sh), 0.0f);
      sB[afrag(mt, kc2, hi2, m2) + e2] = (_Float16)h;
    }
  }
  __syncthreads();
  for (int i = t; i < CH * CH; i += 256) {
    const int kk = i >> 6, cc = i & 63;
    sW[wfrag(cc >> 4, kk >> 5, (kk >> 3) & 1, cc & 15) + ((kk & 7) | (((kk >> 4) & 1) << 3))] = (_Float16)Ww2[i];
  }
  __syncthreads();

  // ---- GEMM2 + softmax over k + weighted combine, all in registers ----
  #pragma unroll
  for (int i = 0; i < 8; ++i) {
    const int T = wv32 * 8 + i;
    const int mt = T >> 2, nt = T & 3;          // mt == point p; tile rows == k
    const int col = nt * 16 + m16;
    v8f acc = {};
    #pragma unroll
    for (int kc = 0; kc < 2; ++kc) {
      const v16h a = *(const v16h*)&sB[afrag(mt, kc, hi, m16)];
      const v16h b = *(const v16h*)&sW[wfrag(nt, kc, hi, m16)];
      acc = __builtin_amdgcn_wmma_f32_16x16x32_f16(false, a, false, b, (short)0, acc, false, false);
    }
    const float b2 = sBN[5 * CH + col];
    float wvr[8], mx = -1e30f;
    #pragma unroll
    for (int j = 0; j < 8; ++j) { wvr[j] = acc[j] + b2; mx = fmaxf(mx, wvr[j]); }
    mx = fmaxf(mx, __shfl_xor(mx, 16, 32));     // lane pair holds all 16 k's
    float se = 0.0f;
    #pragma unroll
    for (int j = 0; j < 8; ++j) { wvr[j] = __expf(wvr[j] - mx); se += wvr[j]; }
    se += __shfl_xor(se, 16, 32);
    const float inv = 1.0f / se;
    float o = 0.0f;
    #pragma unroll
    for (int j = 0; j < 8; ++j) {
      const int krow = j + (hi << 3);
      o = fmaf(wvr[j] * inv, sV[(mt * 16 + krow) * ROWV + col], o);
    }
    o += __shfl_xor(o, 16, 32);
    if (hi == 0) out[(size_t)(n0 + mt) * CH + col] = o;
  }
}

// ---------------------------------------------------------------------------
extern "C" void kernel_launch(void* const* d_in, const int* in_sizes, int n_in,
                              void* d_out, int out_size, void* d_ws, size_t ws_size,
                              hipStream_t stream) {
  (void)in_sizes; (void)n_in; (void)out_size; (void)ws_size;
  const float* x     = (const float*)d_in[0];
  const float* y     = (const float*)d_in[1];
  const float* y_xyz = (const float*)d_in[2];
  const int*   idx   = (const int*)  d_in[3];
  const float* Wq  = (const float*)d_in[4];  const float* bq  = (const float*)d_in[5];
  const float* Wk  = (const float*)d_in[6];  const float* bk  = (const float*)d_in[7];
  const float* Wv  = (const float*)d_in[8];  const float* bv  = (const float*)d_in[9];
  const float* Wu  = (const float*)d_in[10]; const float* bu  = (const float*)d_in[11];
  const float* gu  = (const float*)d_in[12]; const float* bnu = (const float*)d_in[13];
  const float* Wp1 = (const float*)d_in[14]; const float* bp1 = (const float*)d_in[15];
  const float* gp  = (const float*)d_in[16]; const float* bnp = (const float*)d_in[17];
  const float* Wp2 = (const float*)d_in[18]; const float* bp2 = (const float*)d_in[19];
  const float* Wa1 = (const float*)d_in[20]; const float* ba1 = (const float*)d_in[21];
  const float* ga  = (const float*)d_in[22]; const float* bna = (const float*)d_in[23];
  const float* Wa2 = (const float*)d_in[24]; const float* ba2 = (const float*)d_in[25];
  const float* gw1 = (const float*)d_in[26]; const float* bw1b= (const float*)d_in[27];
  const float* Ww1 = (const float*)d_in[28]; const float* bw1 = (const float*)d_in[29];
  const float* gw2 = (const float*)d_in[30]; const float* bw2b= (const float*)d_in[31];
  const float* Ww2 = (const float*)d_in[32]; const float* bw2 = (const float*)d_in[33];
  const float* a_p = (const float*)d_in[34]; const float* b_p = (const float*)d_in[35];
  const float* d_p = (const float*)d_in[36];
  float* out = (float*)d_out;

  float* xq   = (float*)d_ws;             // NPTS*CH
  float* yk   = xq   + (size_t)NPTS * CH;
  float* yv   = yk   + (size_t)NPTS * CH;
  float* xuni = yv   + (size_t)NPTS * CH; // NPTS
  float* wd   = xuni + NPTS;              // CH*KN

  wl_zero<<<1, 256, 0, stream>>>(wd);
  wl_proj<<<NPTS / 16, 32, 0, stream>>>(x, y, Wq, bq, Wk, bk, Wv, bv, xq, yk, yv);
  wl_uni<<<NPTS / 256, 256, 0, stream>>>(x, Wu, bu, gu, bnu, xuni);
  wl_wd<<<NPTS / 128, 256, 0, stream>>>(xq, yk, xuni, idx, wd);
  wl_wdsm<<<1, 64, 0, stream>>>(wd);
  wl_fused<<<NPTS / 16, 256, 0, stream>>>(
      y, y_xyz, idx, yv, wd,
      Wp1, bp1, gp, bnp, Wp2, bp2, Wa1, ba1, ga, bna, Wa2, ba2,
      gw1, bw1b, Ww1, bw1, gw2, bw2b, Ww2, bw2, a_p, b_p, d_p, out);
}
